// MHSA_73967926771815
// MI455X (gfx1250) — matrix operand order repaired and checked by reading, not compiled, not run on hardware
//
#include <hip/hip_runtime.h>
#include <hip/hip_bf16.h>

// ---------------------------------------------------------------------------
// MHSA for x[8,256,32,32]: LN -> QKV proj -> flash attention -> out proj + res
// All GEMMs via v_wmma_f32_16x16x32_f16 (CDNA5 / gfx1250, wave32).
//
// Layout trick: all A-side f16 buffers (Xn, Q, attention-out T) are stored
// with a self-inverse K-swizzle within each 32-element k block
// (swap [8,16) <-> [16,24)), so a 16x32 A fragment is two contiguous 16-byte
// loads per lane. B-side buffers (weights, K) stay plain (already contiguous).
// V is stored transposed so its B fragments are contiguous too.
// ---------------------------------------------------------------------------

typedef __attribute__((ext_vector_type(16))) _Float16 v16h;
typedef __attribute__((ext_vector_type(8)))  float    v8f;

#define WMMA_F16(a, b, c) \
    __builtin_amdgcn_wmma_f32_16x16x32_f16(false, (a), false, (b), (short)0, (c), false, false)

union AFrag {
    v16h v;
    _Float16 h[16];
    unsigned int u[8];
    uint4 q[2];
};

union F8 {
    float  f[8];
    float4 q[2];
};

static constexpr int D     = 256;
static constexpr int NTOK  = 1024;            // 32*32
static constexpr int BATCH = 8;
static constexpr int PERB  = D * NTOK;        // 262144 elems per batch sample

// K-swizzle within a 32 block: swap [8,16) and [16,24). Self-inverse.
__device__ __forceinline__ int kswz(int k) {
    return (k < 8 || k >= 24) ? k : (k < 16 ? k + 8 : k - 8);
}

// ---------------------------------------------------------------------------
// 1) LayerNorm statistics: one block per batch sample.
// ---------------------------------------------------------------------------
__global__ void ln_stats_kernel(const float* __restrict__ x, float* __restrict__ stats) {
    __shared__ float s1[256], s2[256];
    const int b = blockIdx.x, t = threadIdx.x;
    const float* xb = x + (size_t)b * PERB;
    float s = 0.f, ss = 0.f;
    for (int i = 0; i < 1024; ++i) {
        float v = xb[i * 256 + t];             // coalesced
        s += v; ss += v * v;
    }
    s1[t] = s; s2[t] = ss;
    __syncthreads();
    for (int off = 128; off > 0; off >>= 1) {
        if (t < off) { s1[t] += s1[t + off]; s2[t] += s2[t + off]; }
        __syncthreads();
    }
    if (t == 0) {
        float mean = s1[0] * (1.f / (float)PERB);
        float var  = s2[0] * (1.f / (float)PERB) - mean * mean;
        stats[2 * b]     = mean;
        stats[2 * b + 1] = rsqrtf(var + 1e-5f);
    }
}

// ---------------------------------------------------------------------------
// 2) Normalize + transpose [b][c][hw] -> fp16 [b][n][c], K-swizzled in c.
// ---------------------------------------------------------------------------
__global__ void normalize_kernel(const float* __restrict__ x,
                                 const float* __restrict__ stats,
                                 _Float16* __restrict__ xn) {
    int g = blockIdx.x * blockDim.x + threadIdx.x;        // 0 .. 2M-1
    int b   = g >> 18;
    int rem = g & (PERB - 1);
    int c   = rem >> 10;
    int hw  = rem & 1023;
    float v = (x[g] - stats[2 * b]) * stats[2 * b + 1];
    int cs  = (c & ~31) + kswz(c & 31);
    xn[(size_t)b * PERB + (size_t)hw * 256 + cs] = (_Float16)v;
}

// ---------------------------------------------------------------------------
// 3) f32 -> f16 weight conversion (plain layout; weights are B-side)
// ---------------------------------------------------------------------------
__global__ void f32_to_f16_kernel(const float* __restrict__ src,
                                  _Float16* __restrict__ dst, int n) {
    int i = blockIdx.x * blockDim.x + threadIdx.x;
    if (i < n) dst[i] = (_Float16)src[i];
}

// ---------------------------------------------------------------------------
// 4) WMMA GEMM:  Y[b][n][o] = A[b][n][k] * W[o][k] + bias[o]   (K = 256)
//    A is K-swizzled fp16.  Block: 256 thr = 8 waves, tile 128(M) x 128(N);
//    wave w owns rows [w*16, w*16+16) and 8 accumulator tiles across N.
//    MODE 0: plain f16 out (K buffer)
//    MODE 1: K-swizzled f16 out (Q buffer, A-side of attention)
//    MODE 2: transposed f16 out Vt[b][o][n] (V buffer, B-side of PV)
//    MODE 3: f32 out + raw-view residual (final projection)
// ---------------------------------------------------------------------------
template <int MODE>
__global__ void gemm_kernel(const _Float16* __restrict__ A,
                            const _Float16* __restrict__ W,
                            const float* __restrict__ bias,
                            _Float16* __restrict__ outH,
                            float* __restrict__ outF,
                            const float* __restrict__ resid) {
    const int lane = threadIdx.x & 31;
    const int wv   = threadIdx.x >> 5;
    const int ln16 = lane & 15;
    const int hf   = lane >> 4;
    const int b    = blockIdx.z;
    const int rowBase = blockIdx.x * 128 + wv * 16;
    const int colBase = blockIdx.y * 128;

    const _Float16* arow = A + (size_t)b * (NTOK * 256) + (size_t)(rowBase + ln16) * 256;

    v8f acc[8] = {};

    for (int k0 = 0; k0 < 256; k0 += 32) {
        // A fragment: K-swizzled storage -> contiguous 32B per lane half
        AFrag af;
        const _Float16* ap = arow + k0 + 16 * hf;
        af.q[0] = *(const uint4*)(ap);
        af.q[1] = *(const uint4*)(ap + 8);
#pragma unroll
        for (int nt = 0; nt < 8; ++nt) {
            const int col = colBase + nt * 16 + ln16;
            // B fragment: lane n = ln16, k = 8*hf + e (e<8), 16 + 8*hf + (e-8) (e>=8)
            AFrag bf;
            const _Float16* wp = W + (size_t)col * 256 + k0 + 8 * hf;
            bf.q[0] = *(const uint4*)(wp);
            bf.q[1] = *(const uint4*)(wp + 16);
            acc[nt] = WMMA_F16(af.v, bf.v, acc[nt]);
        }
    }

#pragma unroll
    for (int nt = 0; nt < 8; ++nt) {
        const int col = colBase + nt * 16 + ln16;
        const float bv = bias[col];
#pragma unroll
        for (int r = 0; r < 8; ++r) {
            const int row = rowBase + r + 8 * hf;     // C/D layout: M = r + 8*(lane>=16)
            float v = acc[nt][r] + bv;
            if (MODE == 0) {
                outH[(size_t)b * PERB + (size_t)row * 256 + col] = (_Float16)v;
            } else if (MODE == 1) {
                int cs = (col & ~31) + kswz(col & 31);
                outH[(size_t)b * PERB + (size_t)row * 256 + cs] = (_Float16)v;
            } else if (MODE == 2) {
                outH[(size_t)b * PERB + (size_t)col * 1024 + row] = (_Float16)v;
            } else {
                size_t idx = (size_t)b * PERB + (size_t)row * 256 + col;
                outF[idx] = v + resid[idx];
            }
        }
    }
}

// ---------------------------------------------------------------------------
// 5) Flash attention. grid = (Mtiles=8, heads=8, batch=8), 256 thr = 8 waves.
//    Wave owns 16 query rows; DK=32 -> one WMMA per 16x16 score tile.
//    Per 32-column step: score tile f32 -> LDS -> reload in A-fragment (kappa)
//    order so each lane holds 16 in-row values; row max/sum are in-lane trees
//    plus a single shfl_xor(16). P fragment formed directly in registers.
//    Per-row alpha rescale of O broadcast through a 16-float LDS line.
// ---------------------------------------------------------------------------
__global__ void attention_kernel(const _Float16* __restrict__ Q,  // K-swizzled
                                 const _Float16* __restrict__ K,  // plain
                                 const _Float16* __restrict__ Vt, // [b][dv][n]
                                 _Float16* __restrict__ T) {      // K-swizzled
    __shared__ float sbuf[8][16][32];   // per-wave score tiles (16 KB)
    __shared__ float abuf[8][16];       // per-wave row broadcast (512 B)
    const int lane = threadIdx.x & 31;
    const int wv   = threadIdx.x >> 5;
    const int ln16 = lane & 15;
    const int hf   = lane >> 4;
    const int head = blockIdx.y;
    const int b    = blockIdx.z;
    const int rowBase = blockIdx.x * 128 + wv * 16;
    const size_t bo = (size_t)b * (NTOK * 256) + head * 32;

    // Q fragment (K-swizzled storage -> 2 contiguous loads), scaled by DK^-0.5
    AFrag qf;
    {
        const _Float16* qp = Q + bo + (size_t)(rowBase + ln16) * 256 + 16 * hf;
        qf.q[0] = *(const uint4*)(qp);
        qf.q[1] = *(const uint4*)(qp + 8);
        const _Float16 hs = (_Float16)0.17677669529663687f;   // 32^-0.5
#pragma unroll
        for (int e = 0; e < 16; ++e) qf.h[e] = qf.h[e] * hs;
    }

    v8f o_lo = {}, o_hi = {};
    // per-lane softmax state for query row (rowBase + ln16); both lane halves
    // keep identical copies (combined via shfl_xor 16 each step).
    float m_a = -1e30f, l_a = 0.f;

    for (int j0 = 0; j0 < NTOK; j0 += 32) {
        // K^T fragments for key columns [j0,j0+16), [j0+16,j0+32)
        AFrag kf0, kf1;
        const _Float16* kp0 = K + bo + (size_t)(j0 + ln16) * 256 + 8 * hf;
        const _Float16* kp1 = K + bo + (size_t)(j0 + 16 + ln16) * 256 + 8 * hf;
        kf0.q[0] = *(const uint4*)kp0;  kf0.q[1] = *(const uint4*)(kp0 + 16);
        kf1.q[0] = *(const uint4*)kp1;  kf1.q[1] = *(const uint4*)(kp1 + 16);

        // V fragments from transposed V: row dv, contiguous key columns
        AFrag vf0, vf1;
        const _Float16* vp0 = Vt + (size_t)b * PERB + (size_t)(head * 32 + ln16) * 1024 + j0 + 8 * hf;
        const _Float16* vp1 = Vt + (size_t)b * PERB + (size_t)(head * 32 + 16 + ln16) * 1024 + j0 + 8 * hf;
        vf0.q[0] = *(const uint4*)vp0;  vf0.q[1] = *(const uint4*)(vp0 + 16);
        vf1.q[0] = *(const uint4*)vp1;  vf1.q[1] = *(const uint4*)(vp1 + 16);

        if (j0 + 32 < NTOK)
            __builtin_prefetch(K + bo + (size_t)(j0 + 32 + ln16) * 256, 0, 1);

        v8f z = {};
        v8f s0 = WMMA_F16(qf.v, kf0.v, z);
        v8f s1 = WMMA_F16(qf.v, kf1.v, z);

        // scores -> LDS in natural [row][col] layout
#pragma unroll
        for (int r = 0; r < 8; ++r) {
            sbuf[wv][r + 8 * hf][ln16]      = s0[r];
            sbuf[wv][r + 8 * hf][16 + ln16] = s1[r];
        }
        asm volatile("s_wait_dscnt 0" ::: "memory");

        // reload own row in kappa (A-fragment) order: two 8-float runs
        F8 ra, rb;
        const float* srow = &sbuf[wv][ln16][0];
        ra.q[0] = *(const float4*)(srow + 8 * hf);
        ra.q[1] = *(const float4*)(srow + 8 * hf + 4);
        rb.q[0] = *(const float4*)(srow + 16 + 8 * hf);
        rb.q[1] = *(const float4*)(srow + 16 + 8 * hf + 4);

        // in-lane row max over the lane's 16 values, combine halves with 1 shfl
        float mx = ra.f[0];
#pragma unroll
        for (int e = 1; e < 8; ++e) mx = fmaxf(mx, ra.f[e]);
#pragma unroll
        for (int e = 0; e < 8; ++e) mx = fmaxf(mx, rb.f[e]);
        mx = fmaxf(mx, __shfl_xor(mx, 16, 32));

        float m_new = fmaxf(m_a, mx);
        float alpha = __expf(m_a - m_new);

        AFrag pf;
        float rs = 0.f;
#pragma unroll
        for (int e = 0; e < 8; ++e) {
            float p0 = __expf(ra.f[e] - m_new);
            float p1 = __expf(rb.f[e] - m_new);
            rs += p0 + p1;
            pf.h[e]     = (_Float16)p0;   // already in kappa order
            pf.h[8 + e] = (_Float16)p1;
        }
        rs += __shfl_xor(rs, 16, 32);
        l_a = l_a * alpha + rs;
        m_a = m_new;

        // broadcast alpha from A-lane layout to C/D row layout via LDS line
        abuf[wv][ln16] = alpha;           // both halves write identical value
        asm volatile("s_wait_dscnt 0" ::: "memory");
        F8 av;
        av.q[0] = *(const float4*)&abuf[wv][8 * hf];
        av.q[1] = *(const float4*)&abuf[wv][8 * hf + 4];
#pragma unroll
        for (int r = 0; r < 8; ++r) { o_lo[r] *= av.f[r]; o_hi[r] *= av.f[r]; }

        o_lo = WMMA_F16(pf.v, vf0.v, o_lo);
        o_hi = WMMA_F16(pf.v, vf1.v, o_hi);
    }

    // epilogue: broadcast l to C/D rows, normalize, store K-swizzled fp16
    abuf[wv][ln16] = l_a;
    asm volatile("s_wait_dscnt 0" ::: "memory");
    F8 lv;
    lv.q[0] = *(const float4*)&abuf[wv][8 * hf];
    lv.q[1] = *(const float4*)&abuf[wv][8 * hf + 4];

    _Float16* tb = T + bo;
    const int pos_lo = kswz(ln16);
    const int pos_hi = kswz(16 + ln16);
#pragma unroll
    for (int r = 0; r < 8; ++r) {
        const float inv = 1.0f / lv.f[r];
        const int row = rowBase + r + 8 * hf;
        tb[(size_t)row * 256 + pos_lo] = (_Float16)(o_lo[r] * inv);
        tb[(size_t)row * 256 + pos_hi] = (_Float16)(o_hi[r] * inv);
    }
}

// ---------------------------------------------------------------------------
// Launch
// ---------------------------------------------------------------------------
extern "C" void kernel_launch(void* const* d_in, const int* in_sizes, int n_in,
                              void* d_out, int out_size, void* d_ws, size_t ws_size,
                              hipStream_t stream) {
    const float* x  = (const float*)d_in[0];
    const float* Wq = (const float*)d_in[1];
    const float* bq = (const float*)d_in[2];
    const float* Wk = (const float*)d_in[3];
    const float* bk = (const float*)d_in[4];
    const float* Wv = (const float*)d_in[5];
    const float* bv = (const float*)d_in[6];
    const float* Wo = (const float*)d_in[7];
    const float* bo = (const float*)d_in[8];
    float* out = (float*)d_out;

    char* ws = (char*)d_ws;
    const size_t SZ_XN = (size_t)BATCH * PERB * sizeof(_Float16);   // 4 MB
    const size_t SZ_W  = (size_t)65536 * sizeof(_Float16);          // 128 KB each
    float*    stats = (float*)ws;                                   // 64 B
    _Float16* Xn  = (_Float16*)(ws + 256);                          // K-swizzled
    _Float16* Wqh = (_Float16*)(ws + 256 + SZ_XN);
    _Float16* Wkh = (_Float16*)(ws + 256 + SZ_XN + SZ_W);
    _Float16* Wvh = (_Float16*)(ws + 256 + SZ_XN + 2 * SZ_W);
    _Float16* Woh = (_Float16*)(ws + 256 + SZ_XN + 3 * SZ_W);
    _Float16* Qb  = (_Float16*)(ws + 256 + SZ_XN + 4 * SZ_W);       // K-swizzled
    _Float16* Kb  = (_Float16*)(ws + 256 + 2 * SZ_XN + 4 * SZ_W);   // plain
    _Float16* Vt  = (_Float16*)(ws + 256 + 3 * SZ_XN + 4 * SZ_W);   // transposed
    _Float16* Tb  = (_Float16*)(ws + 256 + 4 * SZ_XN + 4 * SZ_W);   // K-swizzled

    // 1) LN stats + weight conversion (independent)
    ln_stats_kernel<<<BATCH, 256, 0, stream>>>(x, stats);
    f32_to_f16_kernel<<<64, 1024, 0, stream>>>(Wq, Wqh, 65536);
    f32_to_f16_kernel<<<64, 1024, 0, stream>>>(Wk, Wkh, 65536);
    f32_to_f16_kernel<<<64, 1024, 0, stream>>>(Wv, Wvh, 65536);
    f32_to_f16_kernel<<<64, 1024, 0, stream>>>(Wo, Woh, 65536);

    // 2) normalize + transpose (K-swizzled)
    normalize_kernel<<<2048, 1024, 0, stream>>>(x, stats, Xn);

    // 3) Q/K/V projections
    dim3 gg(8, 2, BATCH);
    gemm_kernel<1><<<gg, 256, 0, stream>>>(Xn, Wqh, bq, Qb, nullptr, nullptr);
    gemm_kernel<0><<<gg, 256, 0, stream>>>(Xn, Wkh, bk, Kb, nullptr, nullptr);
    gemm_kernel<2><<<gg, 256, 0, stream>>>(Xn, Wvh, bv, Vt, nullptr, nullptr);

    // 4) flash attention
    attention_kernel<<<dim3(8, 8, BATCH), 256, 0, stream>>>(Qb, Kb, Vt, Tb);

    // 5) output projection + bias + raw-view residual
    gemm_kernel<3><<<gg, 256, 0, stream>>>(Tb, Woh, bo, nullptr, out, x);

    (void)in_sizes; (void)n_in; (void)out_size; (void)ws_size;
}
